// Transformer_7404523618735
// MI455X (gfx1250) — matrix operand order repaired and checked by reading, not compiled, not run on hardware
//
#include <hip/hip_runtime.h>
#include <cstdint>

#define VOCAB  50257
#define DMODEL 512
#define NHEAD  8
#define HSIZE  64
#define BATCH  2
#define SEQ    2048
#define ROWS   (BATCH*SEQ)   // 4096
#define QKVW   1536          // q|k|v packed width

typedef __attribute__((ext_vector_type(16))) _Float16 v16h;
typedef __attribute__((ext_vector_type(8)))  float    v8f;
typedef __attribute__((ext_vector_type(4)))  unsigned u32x4;
typedef __attribute__((ext_vector_type(8)))  int      i32x8;
typedef __attribute__((ext_vector_type(4)))  int      i32x4;

#if defined(__gfx1250__) && __has_builtin(__builtin_amdgcn_tensor_load_to_lds) && \
    __has_builtin(__builtin_amdgcn_s_wait_tensorcnt)
#define HAVE_TDM 1
#endif
#if defined(__gfx1250__) && __has_builtin(__builtin_amdgcn_global_load_async_to_lds_b128)
#define HAVE_ASYNC 1
#endif

typedef __attribute__((address_space(1))) i32x4 as1_v4i;
typedef __attribute__((address_space(3))) i32x4 as3_v4i;

static __device__ __forceinline__ v8f wmma_f16(v16h a, v16h b, v8f c) {
  // D = A(16x32 f16) * B(32x16 f16) + C(16x16 f32)
  return __builtin_amdgcn_wmma_f32_16x16x32_f16(false, a, false, b, (short)0, c,
                                                false, false);
}

static __device__ __forceinline__ void lds_wait() {
  asm volatile("s_wait_dscnt 0x0" ::: "memory");
}

static __device__ __forceinline__ unsigned lds_off32(const void* p) {
  // flat LDS addresses carry the byte offset in the low 32 bits (ISA 10.2)
  return (unsigned)(size_t)p;
}

static __device__ __forceinline__ unsigned h_bits(_Float16 h) {
  union { _Float16 h; unsigned short u; } c; c.h = h; return (unsigned)c.u;
}

#ifdef HAVE_ASYNC
static __device__ __forceinline__ void async_g2l_b128(const void* g, void* l) {
  __builtin_amdgcn_global_load_async_to_lds_b128(
      (as1_v4i*)(size_t)g, (as3_v4i*)(size_t)lds_off32(l), 0, 0);
}
static __device__ __forceinline__ void async_wait() {
#if __has_builtin(__builtin_amdgcn_s_wait_asynccnt)
  __builtin_amdgcn_s_wait_asynccnt(0);
#else
  asm volatile("s_wait_asynccnt 0x0" ::: "memory");
#endif
}
#endif

#ifdef HAVE_TDM
// 2D tile DMA: global (row-major, stride0 elements of 2 bytes) -> LDS packed.
// tensor_d0/d1 are OOB limits relative to the tile start (reads beyond -> 0).
static __device__ __forceinline__ void tdm_load_2d(
    unsigned ldsOff, const void* gaddr,
    unsigned tensor_d0, unsigned tensor_d1,
    unsigned tile_d0, unsigned tile_d1, unsigned stride0)
{
  size_t ga = (size_t)gaddr;
  u32x4 g0;
  g0[0] = 1u;                                            // count=1 (valid, user)
  g0[1] = ldsOff;                                        // lds_addr (bytes)
  g0[2] = (unsigned)(ga & 0xffffffffu);                  // global_addr[31:0]
  g0[3] = (unsigned)((ga >> 32) & 0x01ffffffu)           // global_addr[56:32]
        | (2u << 30);                                    // type=2 ("image")
  i32x8 g1;
  g1[0] = (int)(1u << 16);                               // data_size=1 (2 bytes)
  g1[1] = (int)((tensor_d0 & 0xffffu) << 16);            // tensor_dim0[15:0]
  g1[2] = (int)((tensor_d0 >> 16) & 0xffffu)             // tensor_dim0[31:16]
        | (int)((tensor_d1 & 0xffffu) << 16);            // tensor_dim1[15:0]
  g1[3] = (int)((tensor_d1 >> 16) & 0xffffu)             // tensor_dim1[31:16]
        | (int)((tile_d0 & 0xffffu) << 16);              // tile_dim0
  g1[4] = (int)(tile_d1 & 0xffffu);                      // tile_dim1 (tile_dim2=0)
  g1[5] = (int)stride0;                                  // tensor_dim0_stride[31:0]
  g1[6] = 0;                                             // stride0[47:32], dim1_stride lo
  g1[7] = 0;
  i32x4 gz4 = {0, 0, 0, 0};
  i32x8 gz8 = {0, 0, 0, 0, 0, 0, 0, 0};
  // amdgpu-toolchain (clang-23) 6-arg form: (g0, g1, g2, g3, extra, cpol)
  __builtin_amdgcn_tensor_load_to_lds(g0, g1, gz4, gz4, gz8, 0);
}
#endif

// ---------------------------------------------------------------------------
// x = tok_emb[idx] + pos_emb[t]  ->  f16
// ---------------------------------------------------------------------------
__global__ __launch_bounds__(256) void embed_kernel(
    const int* __restrict__ idx, const float* __restrict__ tok,
    const float* __restrict__ pos, _Float16* __restrict__ xh)
{
  const int row = blockIdx.x;               // 0..ROWS-1
  const int t = row & (SEQ - 1);
  const int token = idx[row];
  for (int d = threadIdx.x; d < DMODEL; d += 256)
    xh[(size_t)row * DMODEL + d] =
        (_Float16)(tok[(size_t)token * DMODEL + d] + pos[(size_t)t * DMODEL + d]);
}

// wq/wk/wv [H,D,HS] -> f16 packed [D, 1536] with col = m*512 + h*64 + e
__global__ __launch_bounds__(256) void pack_qkv_kernel(
    const float* __restrict__ wq, const float* __restrict__ wk,
    const float* __restrict__ wv, _Float16* __restrict__ w)
{
  int i = blockIdx.x * 256 + threadIdx.x;
  if (i >= DMODEL * DMODEL) return;
  int d = i >> 9;
  int c = i & 511;
  int h = c >> 6, e = c & 63;
  size_t src = ((size_t)h * DMODEL + d) * HSIZE + e;
  w[(size_t)d * QKVW + c]          = (_Float16)wq[src];
  w[(size_t)d * QKVW + 512 + c]    = (_Float16)wk[src];
  w[(size_t)d * QKVW + 1024 + c]   = (_Float16)wv[src];
}

__global__ __launch_bounds__(256) void cvt_f16_kernel(
    const float* __restrict__ s, _Float16* __restrict__ d, long n)
{
  long i = (long)blockIdx.x * 256 + threadIdx.x;
  if (i < n) d[i] = (_Float16)s[i];
}

// ---------------------------------------------------------------------------
// Generic f16 WMMA GEMM: C[M,N] = A[M,K] * B[K,N]  (row-major, f32 accum)
// 256 threads = 8 waves; macro tile 128(M) x 64(N); K stepped by 32.
// Tiles staged to LDS by the Tensor Data Mover (TDM) when available.
// ---------------------------------------------------------------------------
template<bool BIAS, bool RELU, bool OUTF16>
__global__ __launch_bounds__(256) void gemm_wmma_kernel(
    const _Float16* __restrict__ A, int lda,
    const _Float16* __restrict__ B, int ldb,
    const float* __restrict__ bias,
    float* __restrict__ Cf, _Float16* __restrict__ Ch, int ldc,
    int M, int N, int K)
{
  __shared__ __align__(16) _Float16 As[128 * 32];   // [m][k]
  __shared__ __align__(16) _Float16 Bs[32 * 64];    // [k][n]
  const int tid  = threadIdx.x;
  const int wave = tid >> 5;
  const int lane = tid & 31;
  const int mBase = blockIdx.y * 128;
  const int nBase = blockIdx.x * 64;
  const int m  = lane & 15;
  const int hi = lane >> 4;      // lane half (ISA layout selector)
  const int n  = m;

  const v8f vzero = {0.f,0.f,0.f,0.f,0.f,0.f,0.f,0.f};
  v8f acc[4];
#pragma unroll
  for (int t = 0; t < 4; ++t) acc[t] = vzero;

#ifdef HAVE_TDM
  const unsigned AsOff = lds_off32(&As[0]);
  const unsigned BsOff = lds_off32(&Bs[0]);
#endif

  for (int k0 = 0; k0 < K; k0 += 32) {
#ifdef HAVE_TDM
    if (wave == 0) {        // A tile: 128 rows x 32 k
      tdm_load_2d(AsOff, &A[(size_t)mBase * lda + k0],
                  (unsigned)(K - k0), 128u, 32u, 128u, (unsigned)lda);
    } else if (wave == 1) { // B tile: 32 k x 64 n (zero-pad past N via OOB)
      tdm_load_2d(BsOff, &B[(size_t)k0 * ldb + nBase],
                  (unsigned)(N - nBase), 32u, 64u, 32u, (unsigned)ldb);
    }
    if (wave < 2) __builtin_amdgcn_s_wait_tensorcnt(0);
#else
    // stage A 128x32 (row-major, 8-half vector copies)
    for (int i = tid; i < 512; i += 256) {
      int r = i >> 2, s = i & 3;
      *(uint4*)&As[r * 32 + s * 8] =
          *(const uint4*)&A[(size_t)(mBase + r) * lda + k0 + s * 8];
    }
    // stage B 32x64 row-major, zero-pad n >= N
    {
      int kr = tid >> 3, s = tid & 7;
      size_t gb = (size_t)(k0 + kr) * ldb + nBase + s * 8;
#pragma unroll
      for (int j = 0; j < 8; ++j) {
        int nG = nBase + s * 8 + j;
        _Float16 val = (_Float16)0.f;
        if (nG < N) val = B[gb + j];
        Bs[kr * 64 + s * 8 + j] = val;
      }
    }
#endif
    __syncthreads();

    // A fragment (16x32 f16 layout): VGPR v holds K pair {k,k+1}
    union { v16h v; unsigned u[8]; } ua;
#pragma unroll
    for (int vv = 0; vv < 8; ++vv) {
      int k = (hi << 4) + (vv << 1);
      ua.u[vv] = *(const unsigned*)&As[(wave * 16 + m) * 32 + k];
    }
    const unsigned short* Bsu = (const unsigned short*)Bs;
#pragma unroll
    for (int t = 0; t < 4; ++t) {
      union { v16h v; unsigned u[8]; } ub;   // B 32x16 layout: lane=n, k=16*hi+2v
#pragma unroll
      for (int vv = 0; vv < 8; ++vv) {
        int k = (hi << 4) + (vv << 1);
        unsigned lo  = Bsu[k * 64 + t * 16 + n];
        unsigned hiw = Bsu[(k + 1) * 64 + t * 16 + n];
        ub.u[vv] = lo | (hiw << 16);
      }
      acc[t] = wmma_f16(ua.v, ub.v, acc[t]);
    }
    __syncthreads();
  }

  // epilogue: C/D layout m = v + 8*hi, lane = n
#pragma unroll
  for (int t = 0; t < 4; ++t) {
#pragma unroll
    for (int vv = 0; vv < 8; ++vv) {
      int row = mBase + wave * 16 + vv + hi * 8;
      int col = nBase + t * 16 + n;
      if (col < N) {
        float val = acc[t][vv];
        if (BIAS) val += bias[col];
        if (RELU) val = fmaxf(val, 0.f);
        if (OUTF16) Ch[(size_t)row * ldc + col] = (_Float16)val;
        else        Cf[(size_t)row * ldc + col] = val;
      }
    }
  }
}

// ---------------------------------------------------------------------------
// Flash attention: one wave per 16-row Q tile of one (b,h).
// qkv: [ROWS][1536] f16 with q|k|v at col offsets 0|512|1024, per-head +h*64.
// ---------------------------------------------------------------------------
__global__ __launch_bounds__(128) void attention_kernel(
    const _Float16* __restrict__ qkv, _Float16* __restrict__ att)
{
  __shared__ __align__(16) _Float16 Qs[4][16 * 64];
  __shared__ __align__(16) _Float16 Ps[4][16 * 32];
  const int wave = threadIdx.x >> 5;
  const int lane = threadIdx.x & 31;
  const int tile = blockIdx.x * 4 + wave;   // 0..2047
  const int qt = tile & 127;                // T/16 tiles
  const int bh = tile >> 7;
  const int h = bh & 7;
  const int b = bh >> 3;
  const int t0 = qt * 16;
  const int m  = lane & 15;
  const int hi = lane >> 4;
  const int n  = m;

  // stage Q tile [16][64] into per-wave LDS region (async when available)
#pragma unroll
  for (int i = lane; i < 128; i += 32) {
    int r = i >> 3, s = i & 7;
#ifdef HAVE_ASYNC
    async_g2l_b128(&qkv[(size_t)(b * SEQ + t0 + r) * QKVW + h * 64 + s * 8],
                   &Qs[wave][r * 64 + s * 8]);
#else
    *(uint4*)&Qs[wave][r * 64 + s * 8] =
        *(const uint4*)&qkv[(size_t)(b * SEQ + t0 + r) * QKVW + h * 64 + s * 8];
#endif
  }
#ifdef HAVE_ASYNC
  async_wait();   // per-wave ASYNCcnt: wave-private tile, no barrier needed
#endif
  lds_wait();

  union { v16h v; unsigned u[8]; } qa0, qa1;   // Q as A-frag, K-chunks e=0..31, 32..63
#pragma unroll
  for (int vv = 0; vv < 8; ++vv) {
    int k = (hi << 4) + (vv << 1);
    qa0.u[vv] = *(const unsigned*)&Qs[wave][m * 64 + k];
    qa1.u[vv] = *(const unsigned*)&Qs[wave][m * 64 + 32 + k];
  }

  const v8f vzero = {0.f,0.f,0.f,0.f,0.f,0.f,0.f,0.f};
  float mstat[8], lstat[8];
  v8f O[4];
#pragma unroll
  for (int vv = 0; vv < 8; ++vv) { mstat[vv] = -1e30f; lstat[vv] = 0.f; }
#pragma unroll
  for (int t = 0; t < 4; ++t) O[t] = vzero;

  const int kbMax = (t0 + 15) >> 5;
  for (int kb = 0; kb <= kbMax; ++kb) {
    const int j0 = kb * 32;
    v8f S[2];
#pragma unroll
    for (int s = 0; s < 2; ++s) {
      const int key = j0 + s * 16 + n;
      const size_t kRow = (size_t)(b * SEQ + key) * QKVW + 512 + h * 64;
      union { v16h v; unsigned u[8]; } b0, b1;   // K^T as B-frag (e along K-dim)
#pragma unroll
      for (int vv = 0; vv < 8; ++vv) {
        int kd = (hi << 4) + (vv << 1);
        b0.u[vv] = *(const unsigned*)&qkv[kRow + kd];
        b1.u[vv] = *(const unsigned*)&qkv[kRow + 32 + kd];
      }
      v8f z = vzero;
      z = wmma_f16(qa0.v, b0.v, z);
      z = wmma_f16(qa1.v, b1.v, z);
      S[s] = z;
    }
    // online softmax (rows live in 16-lane halves; m = v + 8*hi)
#pragma unroll
    for (int vv = 0; vv < 8; ++vv) {
      const int row = t0 + vv + hi * 8;
      float a0 = S[0][vv] * 0.125f;    // 1/sqrt(64)
      float a1 = S[1][vv] * 0.125f;
      if (j0 + n > row)      a0 = -1e30f;
      if (j0 + 16 + n > row) a1 = -1e30f;
      float mx = fmaxf(a0, a1);
#pragma unroll
      for (int off = 1; off < 16; off <<= 1)
        mx = fmaxf(mx, __shfl_xor(mx, off, 32));
      const float mnew = fmaxf(mstat[vv], mx);
      const float alpha = __expf(mstat[vv] - mnew);
      const float p0 = __expf(a0 - mnew);
      const float p1 = __expf(a1 - mnew);
      float rs = p0 + p1;
#pragma unroll
      for (int off = 1; off < 16; off <<= 1)
        rs += __shfl_xor(rs, off, 32);
      lstat[vv] = lstat[vv] * alpha + rs;
      mstat[vv] = mnew;
#pragma unroll
      for (int t = 0; t < 4; ++t) O[t][vv] *= alpha;
      Ps[wave][(vv + hi * 8) * 32 + n]      = (_Float16)p0;
      Ps[wave][(vv + hi * 8) * 32 + 16 + n] = (_Float16)p1;
    }
    lds_wait();
    union { v16h v; unsigned u[8]; } pa;   // P as A-frag 16x32
#pragma unroll
    for (int vv = 0; vv < 8; ++vv) {
      int k = (hi << 4) + (vv << 1);
      pa.u[vv] = *(const unsigned*)&Ps[wave][m * 32 + k];
    }
    const size_t vBase = (size_t)(b * SEQ + j0) * QKVW + 1024 + h * 64;
#pragma unroll
    for (int eb = 0; eb < 4; ++eb) {
      union { v16h v; unsigned u[8]; } vb;   // V as B-frag: k = key-local
#pragma unroll
      for (int vv = 0; vv < 8; ++vv) {
        int kk = (hi << 4) + (vv << 1);
        unsigned lo  = h_bits(qkv[vBase + (size_t)kk * QKVW + eb * 16 + n]);
        unsigned hiw = h_bits(qkv[vBase + (size_t)(kk + 1) * QKVW + eb * 16 + n]);
        vb.u[vv] = lo | (hiw << 16);
      }
      O[eb] = wmma_f16(pa.v, vb.v, O[eb]);
    }
    lds_wait();   // keep Ps reads ordered before next iteration's writes
  }

  // normalize + store att f16 (head-concat layout [row][h*64+e])
#pragma unroll
  for (int vv = 0; vv < 8; ++vv) {
    const float inv = 1.0f / lstat[vv];
    const int row = b * SEQ + t0 + vv + hi * 8;
#pragma unroll
    for (int eb = 0; eb < 4; ++eb)
      att[(size_t)row * DMODEL + h * 64 + eb * 16 + n] =
          (_Float16)(O[eb][vv] * inv);
  }
}

// ---------------------------------------------------------------------------
// Cross-entropy: per-row logsumexp - logit[target], then mean.
// ---------------------------------------------------------------------------
__global__ __launch_bounds__(256) void row_loss_kernel(
    const float* __restrict__ logits, const int* __restrict__ targets,
    float* __restrict__ rowNll)
{
  __shared__ float red[256];
  const int row = blockIdx.x;
  const float* lp = logits + (size_t)row * VOCAB;
  float mx = -1e30f;
  for (int i = threadIdx.x; i < VOCAB; i += 256) mx = fmaxf(mx, lp[i]);
  red[threadIdx.x] = mx; __syncthreads();
  for (int s = 128; s > 0; s >>= 1) {
    if (threadIdx.x < s) red[threadIdx.x] = fmaxf(red[threadIdx.x], red[threadIdx.x + s]);
    __syncthreads();
  }
  mx = red[0]; __syncthreads();
  float sum = 0.f;
  for (int i = threadIdx.x; i < VOCAB; i += 256) sum += __expf(lp[i] - mx);
  red[threadIdx.x] = sum; __syncthreads();
  for (int s = 128; s > 0; s >>= 1) {
    if (threadIdx.x < s) red[threadIdx.x] += red[threadIdx.x + s];
    __syncthreads();
  }
  if (threadIdx.x == 0)
    rowNll[row] = (logf(red[0]) + mx) - lp[targets[row]];
}

__global__ __launch_bounds__(256) void final_loss_kernel(
    const float* __restrict__ rowNll, float* __restrict__ lossOut)
{
  __shared__ float red[256];
  float s = 0.f;
  for (int i = threadIdx.x; i < ROWS; i += 256) s += rowNll[i];
  red[threadIdx.x] = s; __syncthreads();
  for (int k = 128; k > 0; k >>= 1) {
    if (threadIdx.x < k) red[threadIdx.x] += red[threadIdx.x + k];
    __syncthreads();
  }
  if (threadIdx.x == 0) lossOut[0] = red[0] * (1.0f / (float)ROWS);
}

// ---------------------------------------------------------------------------
extern "C" void kernel_launch(void* const* d_in, const int* in_sizes, int n_in,
                              void* d_out, int out_size, void* d_ws, size_t ws_size,
                              hipStream_t stream)
{
  const int*   idx     = (const int*)  d_in[0];
  const int*   targets = (const int*)  d_in[1];
  const float* tok_emb = (const float*)d_in[2];
  const float* pos_emb = (const float*)d_in[3];
  const float* wq      = (const float*)d_in[4];
  const float* wk      = (const float*)d_in[5];
  const float* wv      = (const float*)d_in[6];
  const float* w1      = (const float*)d_in[7];
  const float* b1      = (const float*)d_in[8];
  const float* w_head  = (const float*)d_in[9];
  const float* b_head  = (const float*)d_in[10];

  float* logits  = (float*)d_out;
  float* lossOut = logits + (size_t)ROWS * VOCAB;

  char* ws = (char*)d_ws;
  auto alloc = [&](size_t bytes) -> char* {
    char* p = ws;
    ws += (bytes + 255) & ~(size_t)255;
    return p;
  };
  _Float16* xh     = (_Float16*)alloc((size_t)ROWS * DMODEL * 2);
  _Float16* wqkvh  = (_Float16*)alloc((size_t)DMODEL * QKVW * 2);
  _Float16* w1h    = (_Float16*)alloc((size_t)DMODEL * DMODEL * 2);
  _Float16* wheadh = (_Float16*)alloc((size_t)DMODEL * VOCAB * 2);
  _Float16* qkvh   = (_Float16*)alloc((size_t)ROWS * QKVW * 2);
  _Float16* atth   = (_Float16*)alloc((size_t)ROWS * DMODEL * 2);
  _Float16* xffh   = (_Float16*)alloc((size_t)ROWS * DMODEL * 2);
  float*    rowNll = (float*)   alloc((size_t)ROWS * 4);

  embed_kernel<<<ROWS, 256, 0, stream>>>(idx, tok_emb, pos_emb, xh);
  pack_qkv_kernel<<<(DMODEL * DMODEL + 255) / 256, 256, 0, stream>>>(wq, wk, wv, wqkvh);
  cvt_f16_kernel<<<(DMODEL * DMODEL + 255) / 256, 256, 0, stream>>>(w1, w1h, (long)DMODEL * DMODEL);
  cvt_f16_kernel<<<(unsigned)(((long)DMODEL * VOCAB + 255) / 256), 256, 0, stream>>>(
      w_head, wheadh, (long)DMODEL * VOCAB);

  // QKV: [4096,512] x [512,1536] -> f16
  gemm_wmma_kernel<false,false,true><<<dim3(QKVW / 64, ROWS / 128), 256, 0, stream>>>(
      xh, DMODEL, wqkvh, QKVW, nullptr, nullptr, qkvh, QKVW, ROWS, QKVW, DMODEL);

  attention_kernel<<<(BATCH * NHEAD * (SEQ / 16)) / 4, 128, 0, stream>>>(qkvh, atth);

  // FF: relu(att @ w1 + b1) -> f16
  gemm_wmma_kernel<true,true,true><<<dim3(DMODEL / 64, ROWS / 128), 256, 0, stream>>>(
      atth, DMODEL, w1h, DMODEL, b1, nullptr, xffh, DMODEL, ROWS, DMODEL, DMODEL);

  // Head: x_ff @ w_head + b_head -> f32 logits
  gemm_wmma_kernel<true,false,false><<<dim3((VOCAB + 63) / 64, ROWS / 128), 256, 0, stream>>>(
      xffh, DMODEL, wheadh, VOCAB, b_head, logits, nullptr, VOCAB, ROWS, VOCAB, DMODEL);

  row_loss_kernel<<<ROWS, 256, 0, stream>>>(logits, targets, rowNll);
  final_loss_kernel<<<1, 256, 0, stream>>>(rowNll, lossOut);

  (void)in_sizes; (void)n_in; (void)out_size; (void)ws_size;
}
